// Encoder_13099650253103
// MI455X (gfx1250) — matrix operand order repaired and checked by reading, not compiled, not run on hardware
//
#include <hip/hip_runtime.h>
#include <hip/hip_bf16.h>
#include <math.h>

#define DM 512
#define DFF 2048
#define NHEADS 8
#define DK 64

typedef __attribute__((ext_vector_type(16))) _Float16 v16h;
typedef __attribute__((ext_vector_type(8)))  _Float16 v8h;
typedef __attribute__((ext_vector_type(8)))  float    v8f;
typedef __attribute__((ext_vector_type(4)))  float    fx4;

typedef __attribute__((ext_vector_type(4))) unsigned int u32x4;
typedef __attribute__((ext_vector_type(8))) int          i32x8;
typedef __attribute__((ext_vector_type(4))) int          i32x4;

typedef __attribute__((address_space(3))) char lds_char_t;

static __device__ __forceinline__ v16h pack16(v8h lo, v8h hi) {
    return __builtin_shufflevector(lo, hi, 0,1,2,3,4,5,6,7,8,9,10,11,12,13,14,15);
}

// Convert two 8-float LDS chunks into one f16 A-fragment
static __device__ __forceinline__ v16h cvt8_8(const float* p, const float* q) {
    v16h r;
#pragma unroll
    for (int j = 0; j < 8; ++j) { r[j] = (_Float16)p[j]; r[8 + j] = (_Float16)q[j]; }
    return r;
}
// Convert 16 consecutive LDS floats into one f16 B-fragment
static __device__ __forceinline__ v16h cvt16(const float* p) {
    v16h r;
#pragma unroll
    for (int j = 0; j < 16; ++j) r[j] = (_Float16)p[j];
    return r;
}

// ---------------------------------------------------------------------------
// Tensor Data Mover: async 2D tile load (64 rows x 32 f32, row stride K elems)
// from global to LDS. D# packing per CDNA5 ISA.
// 6-arg builtin form (amdgpu-toolchain / clang-23).
// ---------------------------------------------------------------------------
static __device__ __forceinline__ void tdm_load_tile_64x32(const void* lds_ptr,
                                                           const float* gptr,
                                                           int row_stride_elems)
{
    const unsigned lds_off =
        (unsigned)(unsigned long long)(lds_char_t*)lds_ptr;     // LDS byte offset
    const unsigned long long ga = (unsigned long long)gptr;

    u32x4 g0;
    g0[0] = 1u;                                       // count=1 (valid descriptor)
    g0[1] = lds_off;                                  // lds_addr [63:32]
    g0[2] = (unsigned)ga;                             // global_addr [95:64]
    g0[3] = (unsigned)((ga >> 32) & 0x01ffffffull)    // global_addr [120:96]
          | (2u << 30);                               // type=2 ("image")

    i32x8 g1;
    g1[0] = (int)(2u << 16);      // workgroup_mask=0, data_size=2 (4 bytes)
    g1[1] = (int)(32u << 16);     // tensor_dim0 = 32 (low 16 bits @ [63:48])
    g1[2] = (int)(64u << 16);     // tensor_dim0 hi=0 | tensor_dim1 = 64 (low16)
    g1[3] = (int)(32u << 16);     // tensor_dim1 hi=0 | tile_dim0 = 32
    g1[4] = 64;                   // tile_dim1 = 64, tile_dim2 = 0 (2D)
    g1[5] = row_stride_elems;     // tensor_dim0_stride [191:160]
    g1[6] = 0;                    // stride hi | tensor_dim1_stride lo
    g1[7] = 0;

    const i32x4 z4 = (i32x4){0, 0, 0, 0};
    const i32x8 z8 = (i32x8){0, 0, 0, 0, 0, 0, 0, 0};
    __builtin_amdgcn_tensor_load_to_lds(g0, g1, z4, z4, z8, 0);
}

// ---------------------------------------------------------------------------
// GEMM: C[M,N] = act(A[M,K] @ W[K,N] + bias[N]).
// Weights arrive PRE-TRANSPOSED: Wt[N][K] fp32, so both A and B tiles are
// contiguous-row 64x32 fp32 tiles fetched by the TDM (double-buffered, issued
// by wave 0, synced with s_wait_tensorcnt + workgroup barrier).
// Block = 256 thr (8 waves), tile 64x64, K-step 32, f16 WMMA w/ f32 accum.
// ---------------------------------------------------------------------------
__global__ __launch_bounds__(256)
void gemm_bias_kernel(const float* __restrict__ A, const float* __restrict__ Wt,
                      const float* __restrict__ bias, float* __restrict__ C,
                      int M, int N, int K, int relu)
{
    __shared__ __align__(16) float Af[2][64][32];   // A tile  [m][k]
    __shared__ __align__(16) float Bf[2][64][32];   // Wt tile [n][k]

    const int tid   = threadIdx.x;
    const int lane  = tid & 31;
    const int wave  = tid >> 5;
    const int mtile = wave & 3;
    const int nt0   = (wave >> 2) * 2;
    const int mbase = blockIdx.y * 64;
    const int nbase = blockIdx.x * 64;

    const float* Abase = A  + (size_t)mbase * K;
    const float* Bbase = Wt + (size_t)nbase * K;

    v8f c0 = {}; v8f c1 = {};

    if (wave == 0) {
        tdm_load_tile_64x32(&Af[0][0][0], Abase, K);
        tdm_load_tile_64x32(&Bf[0][0][0], Bbase, K);
    }

    int buf = 0;
    for (int k0 = 0; k0 < K; k0 += 32) {
        if (wave == 0) {
            if (k0 + 32 < K) {
                tdm_load_tile_64x32(&Af[buf ^ 1][0][0], Abase + k0 + 32, K);
                tdm_load_tile_64x32(&Bf[buf ^ 1][0][0], Bbase + k0 + 32, K);
                __builtin_amdgcn_s_wait_tensorcnt(2);   // current tile's 2 loads done
            } else {
                __builtin_amdgcn_s_wait_tensorcnt(0);
            }
        }
        __syncthreads();

        // A fragment: lanes 0-15 -> K 0..15 ; lanes 16-31 -> 16..31
        const int r   = lane & 15;
        const int akb = (lane < 16) ? 0 : 16;
        const float* arow = &Af[buf][mtile * 16 + r][0];
        v16h a = cvt8_8(arow + akb, arow + akb + 8);

        // B fragments: lane owns column n, 16 consecutive K values
        const int bkb = (lane < 16) ? 0 : 16;
        v16h b0 = cvt16(&Bf[buf][nt0 * 16 + r][bkb]);
        v16h b1 = cvt16(&Bf[buf][(nt0 + 1) * 16 + r][bkb]);

        c0 = __builtin_amdgcn_wmma_f32_16x16x32_f16(false, a, false, b0, (short)0, c0, false, false);
        c1 = __builtin_amdgcn_wmma_f32_16x16x32_f16(false, a, false, b1, (short)0, c1, false, false);
        __syncthreads();
        buf ^= 1;
    }

    // Epilogue: C layout (lanes 0-15: rows 0..7; lanes 16-31: rows 8..15)
    const int col0  = nbase + nt0 * 16 + (lane & 15);
    const int col1  = col0 + 16;
    const int rbase = mbase + mtile * 16 + ((lane < 16) ? 0 : 8);
    const float bia0 = bias[col0];
    const float bia1 = bias[col1];
#pragma unroll
    for (int r = 0; r < 8; ++r) {
        float v0 = c0[r] + bia0;
        float v1 = c1[r] + bia1;
        if (relu) { v0 = fmaxf(v0, 0.f); v1 = fmaxf(v1, 0.f); }
        C[(size_t)(rbase + r) * N + col0] = v0;
        C[(size_t)(rbase + r) * N + col1] = v1;
    }
}

// ---------------------------------------------------------------------------
// Weight transpose: Wt[N][K] = W[K][N], 32x32 tiles, fp32. Run once per layer.
// ---------------------------------------------------------------------------
__global__ __launch_bounds__(256)
void transpose_kernel(const float* __restrict__ W, float* __restrict__ Wt,
                      int K, int N)
{
    __shared__ float tile[32][33];
    const int tx = threadIdx.x & 31;
    const int ty = threadIdx.x >> 5;           // 0..7
    const int nb = blockIdx.x * 32;            // N base
    const int kb = blockIdx.y * 32;            // K base
#pragma unroll
    for (int j = 0; j < 32; j += 8)
        tile[ty + j][tx] = W[(size_t)(kb + ty + j) * N + nb + tx];
    __syncthreads();
#pragma unroll
    for (int j = 0; j < 32; j += 8)
        Wt[(size_t)(nb + ty + j) * K + kb + tx] = tile[tx][ty + j];
}

// ---------------------------------------------------------------------------
// Flash attention: Q,K,V are [B,S,512] with head h at feature offset h*64.
// Block = 128 thr (4 waves); wave handles a 16-query tile; WG shares 32-key
// K/V LDS tiles. Online softmax; O written to [B,S,512] (heads concatenated).
// ---------------------------------------------------------------------------
__global__ __launch_bounds__(128)
void attention_kernel(const float* __restrict__ Q, const float* __restrict__ Kg,
                      const float* __restrict__ V, const int* __restrict__ mask,
                      float* __restrict__ O, int S, int mstride)
{
    __shared__ __align__(16) _Float16 Kt[32][72];      // K tile row-major [key][feat]
    __shared__ __align__(16) _Float16 Vt[64][40];      // V tile transposed [feat][key]
    __shared__ __align__(16) _Float16 Pw[4][16][40];   // per-wave P staging [row][key]

    const int tid  = threadIdx.x;
    const int lane = tid & 31;
    const int wave = tid >> 5;
    const int bh   = blockIdx.y;
    const int b    = bh / NHEADS;
    const int h    = bh % NHEADS;
    const int qbase = blockIdx.x * 64 + wave * 16;

    const size_t bOff = (size_t)b * S * DM;
    const float* Qp   = Q + bOff + (size_t)qbase * DM + h * DK;
    const int* maskp  = mask + b * 2048;

    // Q fragments (A-layout), K-dim = feature (64 -> two frags)
    const int qr  = lane & 15;
    const int akb = (lane < 16) ? 0 : 16;
    v16h a0, a1;
#pragma unroll
    for (int j = 0; j < 8; ++j) {
        a0[j]     = (_Float16)Qp[qr * DM + akb + j];
        a0[8 + j] = (_Float16)Qp[qr * DM + akb + 8 + j];
        a1[j]     = (_Float16)Qp[qr * DM + 32 + akb + j];
        a1[8 + j] = (_Float16)Qp[qr * DM + 32 + akb + 8 + j];
    }

    float mi[8], li[8];
    v8f o0 = {}, o1 = {}, o2 = {}, o3 = {};
#pragma unroll
    for (int r = 0; r < 8; ++r) { mi[r] = -INFINITY; li[r] = 0.f; }

    const int krow = tid >> 2;          // 0..31
    const int kcb  = (tid & 3) * 16;    // 0,16,32,48

    for (int kb0 = 0; kb0 < S; kb0 += 32) {
        // Cooperative K/V tile load (32 keys x 64 feats)
        const float* ksrc = Kg + bOff + (size_t)(kb0 + krow) * DM + h * DK + kcb;
        const float* vsrc = V  + bOff + (size_t)(kb0 + krow) * DM + h * DK + kcb;
        if (kb0 + 32 < S) {                       // prefetch next tile into caches
            __builtin_prefetch(ksrc + (size_t)32 * DM, 0, 0);
            __builtin_prefetch(vsrc + (size_t)32 * DM, 0, 0);
        }
#pragma unroll
        for (int q4 = 0; q4 < 4; ++q4) {
            fx4 kv = *(const fx4*)(ksrc + q4 * 4);
            fx4 vv = *(const fx4*)(vsrc + q4 * 4);
#pragma unroll
            for (int j = 0; j < 4; ++j) {
                Kt[krow][kcb + q4 * 4 + j] = (_Float16)kv[j];
                Vt[kcb + q4 * 4 + j][krow] = (_Float16)vv[j];
            }
        }
        __syncthreads();

        // Two 16-key score subtiles
#pragma unroll
        for (int sub = 0; sub < 2; ++sub) {
            const int kcol = sub * 16 + (lane & 15);
            const int fb   = (lane < 16) ? 0 : 16;
            v16h bk0 = pack16(*(const v8h*)&Kt[kcol][fb],
                              *(const v8h*)&Kt[kcol][fb + 8]);
            v16h bk1 = pack16(*(const v8h*)&Kt[kcol][32 + fb],
                              *(const v8h*)&Kt[kcol][32 + fb + 8]);
            v8f s = {};
            s = __builtin_amdgcn_wmma_f32_16x16x32_f16(false, a0, false, bk0, (short)0, s, false, false);
            s = __builtin_amdgcn_wmma_f32_16x16x32_f16(false, a1, false, bk1, (short)0, s, false, false);

            const int mv = maskp[(kb0 + kcol) * mstride];
            float sv[8];
#pragma unroll
            for (int r = 0; r < 8; ++r) {
                float t = s[r] * 0.125f;           // 1/sqrt(64)
                sv[r] = mv ? t : -1.0e9f;
            }
            // row max over the 16 keys (each row lives in one 16-lane half)
            float tm[8];
#pragma unroll
            for (int r = 0; r < 8; ++r) tm[r] = sv[r];
#pragma unroll
            for (int off = 8; off >= 1; off >>= 1)
#pragma unroll
                for (int r = 0; r < 8; ++r) tm[r] = fmaxf(tm[r], __shfl_xor(tm[r], off, 32));

            float p[8], ts[8];
#pragma unroll
            for (int r = 0; r < 8; ++r) {
                float mnew  = fmaxf(mi[r], tm[r]);
                float scale = __expf(mi[r] - mnew);
                p[r]  = __expf(sv[r] - mnew);
                ts[r] = p[r];
                mi[r] = mnew;
                li[r] *= scale;
                o0[r] *= scale; o1[r] *= scale; o2[r] *= scale; o3[r] *= scale;
            }
#pragma unroll
            for (int off = 8; off >= 1; off >>= 1)
#pragma unroll
                for (int r = 0; r < 8; ++r) ts[r] += __shfl_xor(ts[r], off, 32);
#pragma unroll
            for (int r = 0; r < 8; ++r) li[r] += ts[r];

            // Stage P (C-layout -> row-major LDS; wave-local, LDS is in-order)
            const int prow = (lane < 16) ? 0 : 8;
#pragma unroll
            for (int r = 0; r < 8; ++r)
                Pw[wave][prow + r][kcol] = (_Float16)p[r];
        }

        // P(16x32) @ V(32x64): A-frag from Pw, B-frags from transposed Vt
        const int pr  = lane & 15;
        const int pkb = (lane < 16) ? 0 : 16;
        v16h pa = pack16(*(const v8h*)&Pw[wave][pr][pkb],
                         *(const v8h*)&Pw[wave][pr][pkb + 8]);
        const int fcol = lane & 15;
        const int vkb  = (lane < 16) ? 0 : 16;
        v16h bv0 = pack16(*(const v8h*)&Vt[fcol][vkb],      *(const v8h*)&Vt[fcol][vkb + 8]);
        v16h bv1 = pack16(*(const v8h*)&Vt[16 + fcol][vkb], *(const v8h*)&Vt[16 + fcol][vkb + 8]);
        v16h bv2 = pack16(*(const v8h*)&Vt[32 + fcol][vkb], *(const v8h*)&Vt[32 + fcol][vkb + 8]);
        v16h bv3 = pack16(*(const v8h*)&Vt[48 + fcol][vkb], *(const v8h*)&Vt[48 + fcol][vkb + 8]);
        o0 = __builtin_amdgcn_wmma_f32_16x16x32_f16(false, pa, false, bv0, (short)0, o0, false, false);
        o1 = __builtin_amdgcn_wmma_f32_16x16x32_f16(false, pa, false, bv1, (short)0, o1, false, false);
        o2 = __builtin_amdgcn_wmma_f32_16x16x32_f16(false, pa, false, bv2, (short)0, o2, false, false);
        o3 = __builtin_amdgcn_wmma_f32_16x16x32_f16(false, pa, false, bv3, (short)0, o3, false, false);
        __syncthreads();
    }

    // Write O (normalize by row sums)
    const int rbase = (lane < 16) ? 0 : 8;
    const int ncol  = lane & 15;
    float* Op = O + bOff + h * DK;
#pragma unroll
    for (int r = 0; r < 8; ++r) {
        float inv = 1.0f / li[r];
        size_t ro = (size_t)(qbase + rbase + r) * DM;
        Op[ro + ncol]      = o0[r] * inv;
        Op[ro + 16 + ncol] = o1[r] * inv;
        Op[ro + 32 + ncol] = o2[r] * inv;
        Op[ro + 48 + ncol] = o3[r] * inv;
    }
}

// ---------------------------------------------------------------------------
// LayerNorm over d=512 with optional fused residual add; matches reference:
// std = sqrt(sum((x-mean)^2)/(d-1)); out = g*(x-mean)/(std+eps)+b
// ---------------------------------------------------------------------------
__global__ __launch_bounds__(128)
void layernorm_kernel(const float* __restrict__ X, const float* __restrict__ R,
                      const float* __restrict__ g, const float* __restrict__ bta,
                      float* __restrict__ out)
{
    __shared__ float red[2][4];
    const int tid  = threadIdx.x;
    const int lane = tid & 31;
    const int wid  = tid >> 5;
    const size_t row = blockIdx.x;
    const float* xr = X + row * DM;
    const float* rr = R ? (R + row * DM) : nullptr;

    float v[4];
#pragma unroll
    for (int j = 0; j < 4; ++j) {
        int e = tid + j * 128;
        v[j] = xr[e] + (rr ? rr[e] : 0.f);
    }
    float s = v[0] + v[1] + v[2] + v[3];
#pragma unroll
    for (int off = 16; off >= 1; off >>= 1) s += __shfl_xor(s, off, 32);
    if (lane == 0) red[0][wid] = s;
    __syncthreads();
    const float mean = (red[0][0] + red[0][1] + red[0][2] + red[0][3]) * (1.0f / DM);

    float ss = 0.f;
#pragma unroll
    for (int j = 0; j < 4; ++j) { float d = v[j] - mean; ss += d * d; }
#pragma unroll
    for (int off = 16; off >= 1; off >>= 1) ss += __shfl_xor(ss, off, 32);
    if (lane == 0) red[1][wid] = ss;
    __syncthreads();
    const float tot = red[1][0] + red[1][1] + red[1][2] + red[1][3];
    const float inv = 1.0f / (sqrtf(tot / (float)(DM - 1)) + 1e-5f);
#pragma unroll
    for (int j = 0; j < 4; ++j) {
        int e = tid + j * 128;
        out[row * DM + e] = g[e] * (v[j] - mean) * inv + bta[e];
    }
}

// ---------------------------------------------------------------------------
extern "C" void kernel_launch(void* const* d_in, const int* in_sizes, int n_in,
                              void* d_out, int out_size, void* d_ws, size_t ws_size,
                              hipStream_t stream)
{
    const float* x_in = (const float*)d_in[0];
    const int*   mask = (const int*)d_in[1];
    const float* fg   = (const float*)d_in[36];
    const float* fb   = (const float*)d_in[37];

    float* ws = (float*)d_ws;
    size_t off = 0;
    float* F  = ws + off; off += (size_t)4096 * 2048;   // FFN intermediate
    float* Xb[3];
    Xb[0] = ws + off; off += (size_t)4096 * 512;
    Xb[1] = ws + off; off += (size_t)4096 * 512;
    Xb[2] = ws + off; off += (size_t)4096 * 512;
    float* BufQ = ws + off; off += (size_t)4096 * 512;
    float* BufK = ws + off; off += (size_t)4096 * 512;
    float* BufV = ws + off; off += (size_t)4096 * 512;
    float* BufA = ws + off; off += (size_t)4096 * 512;
    // Per-layer transposed-weight slots (reused each layer)
    float* WtQ = ws + off; off += (size_t)DM * DM;
    float* WtK = ws + off; off += (size_t)DM * DM;
    float* WtV = ws + off; off += (size_t)DM * DM;
    float* WtO = ws + off; off += (size_t)DM * DM;
    float* Wt1 = ws + off; off += (size_t)DM * DFF;     // [N=2048][K=512]
    float* Wt2 = ws + off; off += (size_t)DFF * DM;     // [N=512][K=2048]
    float* WtR = ws + off; off += (size_t)DM * (DM / 2);

    auto gemm = [&](const float* A, const float* Wt, const float* bi, float* C,
                    int M, int N, int K, int relu) {
        dim3 grid(N / 64, M / 64);
        gemm_bias_kernel<<<grid, 256, 0, stream>>>(A, Wt, bi, C, M, N, K, relu);
    };
    auto transpose = [&](const float* W, float* Wt, int K, int N) {
        dim3 grid(N / 32, K / 32);
        transpose_kernel<<<grid, 256, 0, stream>>>(W, Wt, K, N);
    };

    const int B = 2;
    int S = 2048, mstride = 1;
    const float* xcur = x_in;
    int xidx = -1;

    for (int phase = 0; phase < 2; ++phase) {
        const int base = phase ? 18 : 2;
        const float* Wq  = (const float*)d_in[base + 0];
        const float* bq  = (const float*)d_in[base + 1];
        const float* Wk  = (const float*)d_in[base + 2];
        const float* bk  = (const float*)d_in[base + 3];
        const float* Wv  = (const float*)d_in[base + 4];
        const float* bv  = (const float*)d_in[base + 5];
        const float* Wo  = (const float*)d_in[base + 6];
        const float* bo  = (const float*)d_in[base + 7];
        const float* W1  = (const float*)d_in[base + 8];
        const float* b1  = (const float*)d_in[base + 9];
        const float* W2  = (const float*)d_in[base + 10];
        const float* b2  = (const float*)d_in[base + 11];
        const float* g1  = (const float*)d_in[base + 12];
        const float* n1b = (const float*)d_in[base + 13];
        const float* g2  = (const float*)d_in[base + 14];
        const float* n2b = (const float*)d_in[base + 15];
        const float* Wr  = phase ? (const float*)d_in[base + 16] : nullptr;
        const float* br  = phase ? (const float*)d_in[base + 17] : nullptr;

        for (int i = 0; i < 3; ++i) {
            const int tokens = B * S;

            // Pre-transpose this layer's weights for the TDM-fed GEMM
            transpose(Wq + (size_t)i * DM * DM,  WtQ, DM, DM);
            transpose(Wk + (size_t)i * DM * DM,  WtK, DM, DM);
            transpose(Wv + (size_t)i * DM * DM,  WtV, DM, DM);
            transpose(Wo + (size_t)i * DM * DM,  WtO, DM, DM);
            transpose(W1 + (size_t)i * DM * DFF, Wt1, DM, DFF);
            transpose(W2 + (size_t)i * DFF * DM, Wt2, DFF, DM);
            if (phase)
                transpose(Wr + (size_t)i * DM * (DM / 2), WtR, DM, DM / 2);

            gemm(xcur, WtQ, bq + (size_t)i * DM, BufQ, tokens, DM, DM, 0);
            gemm(xcur, WtK, bk + (size_t)i * DM, BufK, tokens, DM, DM, 0);
            gemm(xcur, WtV, bv + (size_t)i * DM, BufV, tokens, DM, DM, 0);

            attention_kernel<<<dim3(S / 64, B * NHEADS), 128, 0, stream>>>(
                BufQ, BufK, BufV, mask, BufA, S, mstride);

            gemm(BufA, WtO, bo + (size_t)i * DM, BufQ, tokens, DM, DM, 0);

            // T = LN(x + attn_out) -> BufK
            layernorm_kernel<<<tokens, 128, 0, stream>>>(
                BufQ, xcur, g1 + (size_t)i * DM, n1b + (size_t)i * DM, BufK);

            gemm(BufK, Wt1, b1 + (size_t)i * DFF, F, tokens, DFF, DM, 1);
            gemm(F, Wt2, b2 + (size_t)i * DM, BufV, tokens, DM, DFF, 0);

            const int outIdx = (xidx < 0) ? 0 : (xidx + 1) % 3;
            layernorm_kernel<<<tokens, 128, 0, stream>>>(
                BufK, BufV, g2 + (size_t)i * DM, n2b + (size_t)i * DM, Xb[outIdx]);
            xcur = Xb[outIdx]; xidx = outIdx;

            if (phase) {
                // y = x @ Wr + br : [tokens,512]x[512,256]; reshape (B,S,256)->(B,S/2,512)
                // is a flat-memory no-op, so just halve S and double the mask stride.
                const int yIdx = (xidx + 1) % 3;
                gemm(xcur, WtR, br + (size_t)i * (DM / 2), Xb[yIdx], tokens, DM / 2, DM, 0);
                xcur = Xb[yIdx]; xidx = yIdx;
                S /= 2; mstride *= 2;
            }
        }
    }

    // Final layernorm -> d_out  (tokens = 2*256 = 512)
    layernorm_kernel<<<B * S, 128, 0, stream>>>(xcur, nullptr, fg, fb, (float*)d_out);
}
